// BasicBlock_11759620456943
// MI455X (gfx1250) — 8-bit matrix operand order repaired and checked by reading, compiled, not run on hardware
//
#include <hip/hip_runtime.h>
#include <cstdint>

typedef int v4i __attribute__((ext_vector_type(4)));
typedef int v8i __attribute__((ext_vector_type(8)));

#define Bn   64
#define Cn   128
#define Hn   56
#define Wn   56
#define HP   58
#define WP   58
#define HWn  (Hn*Wn)
#define KTOT 1152
#define XSB  ((size_t)Bn*HP*WP*Cn)
#define WSB  (128*KTOT)

__global__ __launch_bounds__(256)
void k_binarize_x(const float* __restrict__ x, int8_t* __restrict__ xs,
                  int8_t* __restrict__ ms)
{
    int idx = blockIdx.x * blockDim.x + threadIdx.x;
    if (idx >= Bn * HP * WP * (Cn / 4)) return;
    int cq = idx & 31;
    int t  = idx >> 5;
    int wp = t % WP; t /= WP;
    int hp = t % HP; t /= HP;
    int b  = t;
    bool border = (hp == 0) | (hp == HP - 1) | (wp == 0) | (wp == WP - 1);
    uint32_t packed = 0u;
    if (!border) {
        int h = hp - 1, w = wp - 1;
#pragma unroll
        for (int j = 0; j < 4; ++j) {
            int c = cq * 4 + j;
            float v = x[((b * Cn + c) * Hn + h) * Wn + w];
            int8_t s = v > 0.f ? 1 : (v < 0.f ? -1 : 0);
            packed |= (uint32_t)(uint8_t)s << (8 * j);
        }
    }
    int off = ((b * HP + hp) * WP + wp) * Cn + cq * 4;
    *(uint32_t*)(xs + off) = packed;
    if (border) *(uint32_t*)(ms + off) = 0u;
}

__global__ __launch_bounds__(256)
void k_binarize_w(const float* __restrict__ w1, const float* __restrict__ w2,
                  int8_t* __restrict__ z1, int8_t* __restrict__ z2)
{
    int idx = blockIdx.x * blockDim.x + threadIdx.x;
    if (idx >= 2 * 128 * KTOT) return;
    int which = idx / (128 * KTOT);
    int rem   = idx % (128 * KTOT);
    int o   = rem / KTOT;
    int k   = rem % KTOT;
    int tap = k / 128;
    int c   = k % 128;
    const float* w = which ? w2 : w1;
    float v = w[o * KTOT + c * 9 + tap];
    int8_t sgn = v > 0.f ? 1 : (v < 0.f ? -1 : 0);
    int s = c >> 6, rr = c & 63;
    int g = rr >> 4, r16 = rr & 15;
    int hf = r16 >> 3, r8 = r16 & 7;
    int vv = g * 2 + (r8 >> 2);
    int j  = r8 & 3;
    int byteoff = s * 64 + (g & 1) * 32 + (g >> 1) * 16 + r16;
    int8_t* z = which ? z2 : z1;
    z[o * KTOT + tap * 128 + byteoff] = sgn;
}

__global__ void k_bnconst(const float* g1, const float* b1, const float* m1, const float* v1,
                          const float* g2, const float* b2, const float* m2, const float* v2,
                          float* __restrict__ bn)
{
    int t = threadIdx.x;
    if (t < 128) {
        float inv = g1[t] * rsqrtf(v1[t] + 1e-5f);
        bn[t] = inv; bn[128 + t] = b1[t] - m1[t] * inv;
    } else if (t < 256) {
        int c = t - 128;
        float inv = g2[c] * rsqrtf(v2[c] + 1e-5f);
        bn[256 + c] = inv; bn[384 + c] = b2[c] - m2[c] * inv;
    }
}

template <int SECOND>
__global__ __launch_bounds__(256)
void k_conv(const int8_t* __restrict__ xs,
            const int8_t* __restrict__ wsz,
            const float*  __restrict__ bn,
            const float*  __restrict__ xres,
            int8_t* __restrict__ ms,
            float*  __restrict__ out)
{
    int lane  = threadIdx.x & 31;
    int obase = (threadIdx.x >> 5) * 16;
    int n     = lane & 15;
    int half  = lane >> 4;
    int p  = blockIdx.x * 16 + n;
    int b  = p / HWn;
    int hw = p % HWn;
    int h  = hw / Wn, w = hw % Wn;

    v8i acc = {};
#pragma unroll
    for (int tap = 0; tap < 9; ++tap) {
        int r = tap / 3, c = tap % 3;
        const int8_t* brow = xs + ((((size_t)b * HP + h + r) * WP + (w + c)) << 7);
        const int8_t* arow = wsz + (size_t)(obase + n) * KTOT + tap * 128 + half * 32;
#pragma unroll
        for (int s = 0; s < 2; ++s) {
            v8i A = *(const v8i*)(arow + s * 64);
            const int8_t* bq = brow + s * 64 + half * 16;
            v4i blo = *(const v4i*)(bq);
            v4i bhi = *(const v4i*)(bq + 32);
            v8i Bm;
            Bm[0] = blo[0]; Bm[1] = blo[1]; Bm[2] = blo[2]; Bm[3] = blo[3];
            Bm[4] = bhi[0]; Bm[5] = bhi[1]; Bm[6] = bhi[2]; Bm[7] = bhi[3];
            acc = __builtin_amdgcn_wmma_i32_16x16x64_iu8(true, A, true, Bm, acc,
                                                         false, false);
        }
    }

    const float* inv  = bn;
    const float* bias = bn + 128;
    if (SECOND) {
#pragma unroll
        for (int r = 0; r < 8; ++r) {
            int o  = obase + half * 8 + r;
            size_t oi = ((size_t)b * Cn + o) * HWn + hw;
            float v = (float)acc[r] * inv[o] + bias[o] + xres[oi];
            out[oi] = fminf(1.f, fmaxf(-1.f, v));
        }
    } else {
        uint32_t lo = 0, hi = 0;
#pragma unroll
        for (int r = 0; r < 8; ++r) {
            int o = obase + half * 8 + r;
            float v = (float)acc[r] * inv[o] + bias[o];
            int8_t s = v > 0.f ? 1 : (v < 0.f ? -1 : 0);
            if (r < 4) lo |= (uint32_t)(uint8_t)s << (8 * r);
            else       hi |= (uint32_t)(uint8_t)s << (8 * (r - 4));
        }
        size_t off = (((size_t)b * HP + h + 1) * WP + (w + 1)) * Cn + obase + half * 8;
        *(uint32_t*)(ms + off)     = lo;
        *(uint32_t*)(ms + off + 4) = hi;
    }
}

extern "C" void kernel_launch(void* const* d_in, const int* in_sizes, int n_in,
                              void* d_out, int out_size, void* d_ws, size_t ws_size,
                              hipStream_t stream)
{
    const float* x  = (const float*)d_in[0];
    const float* w1 = (const float*)d_in[1];
    const float* w2 = (const float*)d_in[2];
    const float* g1 = (const float*)d_in[3];
    const float* b1 = (const float*)d_in[4];
    const float* m1 = (const float*)d_in[5];
    const float* v1 = (const float*)d_in[6];
    const float* g2 = (const float*)d_in[7];
    const float* b2 = (const float*)d_in[8];
    const float* m2 = (const float*)d_in[9];
    const float* v2 = (const float*)d_in[10];
    float* out = (float*)d_out;

    int8_t* xs = (int8_t*)d_ws;
    int8_t* ms = xs + XSB;
    int8_t* z1 = ms + XSB;
    int8_t* z2 = z1 + WSB;
    float*  bn = (float*)(z2 + WSB);

    k_bnconst<<<1, 256, 0, stream>>>(g1, b1, m1, v1, g2, b2, m2, v2, bn);

    int tb = Bn * HP * WP * (Cn / 4);
    k_binarize_x<<<(tb + 255) / 256, 256, 0, stream>>>(x, xs, ms);

    int tw = 2 * 128 * KTOT;
    k_binarize_w<<<(tw + 255) / 256, 256, 0, stream>>>(w1, w2, z1, z2);

    int tiles = (Bn * HWn) / 16;
    k_conv<0><<<tiles, 256, 0, stream>>>(xs, z1, bn,       nullptr, ms, nullptr);
    k_conv<1><<<tiles, 256, 0, stream>>>(ms, z2, bn + 256, x,  nullptr, out);
}
